// Attention_29128468202113
// MI455X (gfx1250) — matrix operand order repaired and checked by reading, not compiled, not run on hardware
//
#include <hip/hip_runtime.h>
#include <stdint.h>

typedef __attribute__((ext_vector_type(16))) __bf16 v16bf;
typedef __attribute__((ext_vector_type(8)))  float  v8f;
typedef int v4i __attribute__((vector_size(16)));

struct Frag32B { uint4 a; uint4 b; };

__device__ __forceinline__ v16bf ld_frag(const unsigned short* p0, const unsigned short* p1) {
    Frag32B t;
    t.a = *(const uint4*)p0;
    t.b = *(const uint4*)p1;
    return __builtin_bit_cast(v16bf, t);
}

__device__ __forceinline__ unsigned short f32_to_bf16(float f) {
    union { float f; uint32_t u; } c; c.f = f;
    uint32_t u = c.u;
    uint32_t r = u + 0x7FFFu + ((u >> 16) & 1u);   // round-to-nearest-even
    if ((u & 0x7F800000u) == 0x7F800000u) r = u;   // inf/nan passthrough
    return (unsigned short)(r >> 16);
}

__device__ __forceinline__ uint32_t pack_bf16x2(float a, float b) {
    return (uint32_t)f32_to_bf16(a) | ((uint32_t)f32_to_bf16(b) << 16);
}

// ---- CDNA5 async global -> LDS copy (ASYNCcnt path), 16B per lane ----------
typedef __attribute__((address_space(1))) v4i as1_v4i;
typedef __attribute__((address_space(3))) v4i as3_v4i;

__device__ __forceinline__ void async_copy_b128(const unsigned short* gsrc,
                                                unsigned short* ldst) {
#if __has_builtin(__builtin_amdgcn_global_load_async_to_lds_b128)
    __builtin_amdgcn_global_load_async_to_lds_b128(
        (as1_v4i*)(uintptr_t)gsrc,
        (as3_v4i*)(unsigned)(uintptr_t)ldst, 0, 0);
#else
    asm volatile("global_load_async_to_lds_b128 %0, %1, off"
                 :: "v"((unsigned)(uintptr_t)ldst),
                    "v"((unsigned long long)(uintptr_t)gsrc)
                 : "memory");
#endif
}

__device__ __forceinline__ void wait_async0() {
#if __has_builtin(__builtin_amdgcn_s_wait_asynccnt)
    __builtin_amdgcn_s_wait_asynccnt(0);
#else
    asm volatile("s_wait_asynccnt 0" ::: "memory");
#endif
}

// ---------------------------------------------------------------------------
// Elementwise f32 -> bf16 (bandwidth-bound pre-pass; n multiple of 8)
// ---------------------------------------------------------------------------
__global__ __launch_bounds__(256) void cvt_f32_bf16(
    const float* __restrict__ in, unsigned short* __restrict__ out, size_t n)
{
    size_t i = ((size_t)blockIdx.x * 256 + threadIdx.x) * 8;
    if (i >= n) return;
    float4 a = *(const float4*)(in + i);
    float4 b = *(const float4*)(in + i + 4);
    uint4 r;
    r.x = pack_bf16x2(a.x, a.y);
    r.y = pack_bf16x2(a.z, a.w);
    r.z = pack_bf16x2(b.x, b.y);
    r.w = pack_bf16x2(b.z, b.w);
    *(uint4*)(out + i) = r;
}

// ---------------------------------------------------------------------------
// GEMM: C[M,2048] = A[M,2048] * W[2048,2048]^T  (bf16 in, f32 out)
// Block 128x128, BK=32, 8 waves (2x4), wave tile 64x32. Async LDS staging.
// ---------------------------------------------------------------------------
#define DIM  2048
#define LPAD 40   // bf16 elems per LDS row (80B: 16B aligned, conflict-free)

__global__ __launch_bounds__(256) void gemm_bf16(
    const unsigned short* __restrict__ A, const unsigned short* __restrict__ W,
    float* __restrict__ C, int M)
{
    __shared__ __align__(16) unsigned short sA[128 * LPAD];
    __shared__ __align__(16) unsigned short sB[128 * LPAD];

    const int tid  = threadIdx.x;
    const int wid  = tid >> 5;
    const int lane = tid & 31;
    const int g    = lane >> 4;
    const int ln   = lane & 15;

    const int bm = blockIdx.y * 128;
    const int bn = blockIdx.x * 128;
    const int wave_m = wid >> 2;   // 0..1 -> 64 rows
    const int wave_n = wid & 3;    // 0..3 -> 32 cols

    v8f acc[4][2] = {};

    // copy mapping: thread -> (row, 16-elem half); 32B per thread per tile
    const int lrow  = tid >> 1;
    const int lhalf = tid & 1;
    const unsigned short* gA = A + (size_t)(bm + lrow) * DIM + lhalf * 16;
    const unsigned short* gW = W + (size_t)(bn + lrow) * DIM + lhalf * 16;
    unsigned short* dA = sA + lrow * LPAD + lhalf * 16;
    unsigned short* dB = sB + lrow * LPAD + lhalf * 16;

    for (int k0 = 0; k0 < DIM; k0 += 32) {
        async_copy_b128(gA + k0,     dA);
        async_copy_b128(gA + k0 + 8, dA + 8);
        async_copy_b128(gW + k0,     dB);
        async_copy_b128(gW + k0 + 8, dB + 8);
        wait_async0();
        __syncthreads();

        v16bf afr[4];
        #pragma unroll
        for (int mi = 0; mi < 4; mi++) {
            const unsigned short* base = sA + (wave_m * 64 + mi * 16 + ln) * LPAD;
            afr[mi] = ld_frag(base + g * 8, base + 16 + g * 8);
        }
        #pragma unroll
        for (int ni = 0; ni < 2; ni++) {
            const unsigned short* base = sB + (wave_n * 32 + ni * 16 + ln) * LPAD;
            v16bf bfr = ld_frag(base + g * 8, base + 16 + g * 8);
            #pragma unroll
            for (int mi = 0; mi < 4; mi++)
                acc[mi][ni] = __builtin_amdgcn_wmma_f32_16x16x32_bf16(
                    false, afr[mi], false, bfr, (short)0, acc[mi][ni], false, false);
        }
        __syncthreads();
    }

    // epilogue: compile-time row stride -> immediate store offsets
    #pragma unroll
    for (int mi = 0; mi < 4; mi++) {
        #pragma unroll
        for (int ni = 0; ni < 2; ni++) {
            float* cp = C + (size_t)(bm + wave_m * 64 + mi * 16 + g * 8) * DIM
                          + bn + wave_n * 32 + ni * 16 + ln;
            #pragma unroll
            for (int j = 0; j < 8; j++)
                cp[(size_t)j * DIM] = acc[mi][ni][j];
        }
    }
}

// ---------------------------------------------------------------------------
// Per-head RMSNorm + RoPE, f32 (S*B, H*128) -> bf16 (B,H,S,128). One wave/row.
// ---------------------------------------------------------------------------
__global__ __launch_bounds__(256) void rmsnorm_rope(
    const float* __restrict__ in, const float* __restrict__ gamma,
    const float* __restrict__ rope, unsigned short* __restrict__ outbf,
    float oscale, int S, int B, int H)
{
    const int wid  = blockIdx.x * 8 + (threadIdx.x >> 5);
    const int lane = threadIdx.x & 31;
    if (wid >= S * B * H) return;
    const int h = wid % H;
    const int b = (wid / H) % B;
    const int s = wid / (H * B);

    const float* src = in + (size_t)(s * B + b) * (H * 128) + h * 128;

    float4 xv = *(const float4*)(src + lane * 4);
    float ss = xv.x * xv.x + xv.y * xv.y + xv.z * xv.z + xv.w * xv.w;
    #pragma unroll
    for (int m = 16; m >= 1; m >>= 1) ss += __shfl_xor(ss, m, 32);
    const float inv = rsqrtf(ss * (1.0f / 128.0f) + 1e-6f);

    const int i = lane * 2;
    const float x1a = src[i]      * inv * gamma[i];
    const float x1b = src[i + 1]  * inv * gamma[i + 1];
    const float x2a = src[i + 64] * inv * gamma[i + 64];
    const float x2b = src[i + 65] * inv * gamma[i + 65];
    const float fa = rope[(size_t)s * 128 + i];
    const float fb = rope[(size_t)s * 128 + i + 1];
    const float ca = __cosf(fa), sa = __sinf(fa);
    const float cb = __cosf(fb), sb = __sinf(fb);

    unsigned short* dst = outbf + (((size_t)(b * H + h) * S) + s) * 128;
    *(uint32_t*)(dst + i)      = pack_bf16x2((x1a * ca - x2a * sa) * oscale,
                                             (x1b * cb - x2b * sb) * oscale);
    *(uint32_t*)(dst + i + 64) = pack_bf16x2((x1a * sa + x2a * ca) * oscale,
                                             (x1b * sb + x2b * cb) * oscale);
}

// ---------------------------------------------------------------------------
// V: f32 (S*B, H*128) -> bf16 (B,H,128,S)
// ---------------------------------------------------------------------------
__global__ __launch_bounds__(256) void v_transpose_bf16(
    const float* __restrict__ v, unsigned short* __restrict__ vT,
    int S, int B, int H)
{
    size_t idx = (size_t)blockIdx.x * blockDim.x + threadIdx.x;
    if (idx >= (size_t)S * B * H * 128) return;
    const int d = (int)(idx & 127);
    size_t r = idx >> 7;
    const int h = (int)(r % H); r /= H;
    const int b = (int)(r % B);
    const int s = (int)(r / B);
    vT[((size_t)(b * H + h) * 128 + d) * S + s] = f32_to_bf16(v[idx]);
}

// ---------------------------------------------------------------------------
// Flash attention with cooperative async K/V staging in LDS.
// Q,K (B,H,S,128) bf16 (Q pre-scaled by 1/sqrt(Dh)), VT (B,H,128,S) bf16.
// Output bf16 (S,B,H*128) so the Wo GEMM consumes it directly.
// ---------------------------------------------------------------------------
#define KSTRIDE 136  // 272B rows: 16B aligned; 4-bank step -> conflict-free
#define VSTRIDE 40   // 80B rows: 16B aligned; 20-bank step -> conflict-free

__global__ __launch_bounds__(256) void flash_attn(
    const unsigned short* __restrict__ Qb, const unsigned short* __restrict__ Kb,
    const unsigned short* __restrict__ VTb, unsigned short* __restrict__ Obf,
    int S, int B, int H)
{
    __shared__ __align__(16) unsigned short sK[32 * KSTRIDE];
    __shared__ __align__(16) unsigned short sV[128 * VSTRIDE];
    __shared__ __align__(16) unsigned short pbuf[8][16 * 32];

    const int tid  = threadIdx.x;
    const int wid  = tid >> 5;
    const int lane = tid & 31;
    const int g    = lane >> 4;
    const int ln   = lane & 15;

    const int bh = blockIdx.x;           // b*H + h
    const int h  = bh % H;
    const int b  = bh / H;
    const int qbase = blockIdx.y * 128 + wid * 16;

    const unsigned short* Q  = Qb  + (size_t)bh * S * 128;
    const unsigned short* Kp = Kb  + (size_t)bh * S * 128;
    const unsigned short* VT = VTb + (size_t)bh * 128 * S;

    // Q fragments: 16 rows x 128 d, 4 frags of k=32 (A layout)
    v16bf qf[4];
    {
        const unsigned short* qrow = Q + (size_t)(qbase + ln) * 128;
        #pragma unroll
        for (int f = 0; f < 4; f++)
            qf[f] = ld_frag(qrow + f * 32 + g * 8, qrow + f * 32 + 16 + g * 8);
    }

    v8f o[8] = {};
    float mrun[8], lrun[8];
    #pragma unroll
    for (int j = 0; j < 8; j++) { mrun[j] = -1e30f; lrun[j] = 0.0f; }

    unsigned short* pb = pbuf[wid];

    // staging maps: K tile 32x128 (8 threads/row, 16 elems each),
    //               V tile 128x32 (2 threads/row, 16 elems each)
    const int krow = tid >> 3, kpart = tid & 7;
    const int vrow = tid >> 1, vhalf = tid & 1;
    const unsigned short* gK = Kp + (size_t)krow * 128 + kpart * 16;
    const unsigned short* gV = VT + (size_t)vrow * S + vhalf * 16;
    unsigned short* dK = sK + krow * KSTRIDE + kpart * 16;
    unsigned short* dV = sV + vrow * VSTRIDE + vhalf * 16;

    for (int k0 = 0; k0 < S; k0 += 32) {
        async_copy_b128(gK + (size_t)k0 * 128,     dK);
        async_copy_b128(gK + (size_t)k0 * 128 + 8, dK + 8);
        async_copy_b128(gV + k0,                   dV);
        async_copy_b128(gV + k0 + 8,               dV + 8);
        wait_async0();
        __syncthreads();

        // ---- scores: two 16x16 tiles over d=128 ----
        v8f st[2] = {};
        #pragma unroll
        for (int t = 0; t < 2; t++) {
            const unsigned short* kb = sK + (t * 16 + ln) * KSTRIDE;
            #pragma unroll
            for (int f = 0; f < 4; f++) {
                v16bf kf = ld_frag(kb + f * 32 + g * 8, kb + f * 32 + 16 + g * 8);
                st[t] = __builtin_amdgcn_wmma_f32_16x16x32_bf16(
                    false, qf[f], false, kf, (short)0, st[t], false, false);
            }
        }

        // ---- online softmax (row stats replicated across 16-lane groups) ----
        float alpha[8];
        #pragma unroll
        for (int j = 0; j < 8; j++) {
            float v = fmaxf(st[0][j], st[1][j]);
            v = fmaxf(v, __shfl_xor(v, 1, 32));
            v = fmaxf(v, __shfl_xor(v, 2, 32));
            v = fmaxf(v, __shfl_xor(v, 4, 32));
            v = fmaxf(v, __shfl_xor(v, 8, 32));
            const float mnew = fmaxf(mrun[j], v);
            alpha[j] = __expf(mrun[j] - mnew);
            const float p0 = __expf(st[0][j] - mnew);
            const float p1 = __expf(st[1][j] - mnew);
            st[0][j] = p0; st[1][j] = p1;
            float rs = p0 + p1;
            rs += __shfl_xor(rs, 1, 32);
            rs += __shfl_xor(rs, 2, 32);
            rs += __shfl_xor(rs, 4, 32);
            rs += __shfl_xor(rs, 8, 32);
            lrun[j] = lrun[j] * alpha[j] + rs;
            mrun[j] = mnew;
        }
        #pragma unroll
        for (int nt = 0; nt < 8; nt++)
            #pragma unroll
            for (int j = 0; j < 8; j++)
                o[nt][j] *= alpha[j];

        // ---- C-layout -> A-layout reshape of P through per-wave LDS ----
        #pragma unroll
        for (int j = 0; j < 8; j++) {
            const int row = g * 8 + j;
            pb[row * 32 + ln]      = f32_to_bf16(st[0][j]);
            pb[row * 32 + 16 + ln] = f32_to_bf16(st[1][j]);
        }
        asm volatile("s_wait_dscnt 0" ::: "memory");
        v16bf pf;
        {
            const unsigned short* prow = pb + ln * 32;
            pf = ld_frag(prow + g * 8, prow + 16 + g * 8);
        }

        // ---- O += P(16x32) * V(32x128), 8 n-tiles ----
        #pragma unroll
        for (int nt = 0; nt < 8; nt++) {
            const unsigned short* vb = sV + (nt * 16 + ln) * VSTRIDE;
            v16bf vf = ld_frag(vb + g * 8, vb + 16 + g * 8);
            o[nt] = __builtin_amdgcn_wmma_f32_16x16x32_bf16(
                false, pf, false, vf, (short)0, o[nt], false, false);
        }
        __syncthreads();   // protect sK/sV before next iteration's staging
    }

    // ---- epilogue: normalize, bf16, store (S,B,H*128) ----
    #pragma unroll
    for (int nt = 0; nt < 8; nt++) {
        unsigned short* op = Obf + ((size_t)(qbase + g * 8) * B + b) * (size_t)DIM
                                 + h * 128 + nt * 16 + ln;
        #pragma unroll
        for (int j = 0; j < 8; j++)
            op[(size_t)j * B * DIM] = f32_to_bf16(o[nt][j] / lrun[j]);
    }
}

// ---------------------------------------------------------------------------
extern "C" void kernel_launch(void* const* d_in, const int* in_sizes, int n_in,
                              void* d_out, int out_size, void* d_ws, size_t ws_size,
                              hipStream_t stream) {
    (void)in_sizes; (void)n_in; (void)out_size;
    const int S = 2048, B = 2, H = 16;
    const int M = S * B;                      // 4096 rows

    const float* x    = (const float*)d_in[0];
    const float* rope = (const float*)d_in[1];
    const float* Wq   = (const float*)d_in[2];
    const float* Wk   = (const float*)d_in[3];
    const float* Wv   = (const float*)d_in[4];
    const float* Wo   = (const float*)d_in[5];
    const float* gq   = (const float*)d_in[6];
    const float* gk   = (const float*)d_in[7];
    float* out = (float*)d_out;

    // workspace layout
    char* ws = (char*)d_ws;
    const size_t f32_mat = (size_t)M * DIM * sizeof(float);           // 33.5 MB
    const size_t bf_act  = (size_t)M * DIM * sizeof(unsigned short);  // 16.8 MB
    const size_t bf_w    = (size_t)DIM * DIM * sizeof(unsigned short);// 8.4 MB
    unsigned short* x_bf  = (unsigned short*)(ws);
    unsigned short* wq_bf = (unsigned short*)(ws + bf_act);
    unsigned short* wk_bf = (unsigned short*)(ws + bf_act + bf_w);
    unsigned short* wv_bf = (unsigned short*)(ws + bf_act + 2 * bf_w);
    unsigned short* wo_bf = (unsigned short*)(ws + bf_act + 3 * bf_w);
    char* p = ws + bf_act + 4 * bf_w;
    float* q_f32 = (float*)(p);
    float* k_f32 = (float*)(p + f32_mat);
    float* v_f32 = (float*)(p + 2 * f32_mat);
    unsigned short* q_bf = (unsigned short*)(p + 3 * f32_mat);
    unsigned short* k_bf = (unsigned short*)(p + 3 * f32_mat + bf_act);
    unsigned short* vT   = (unsigned short*)(p + 3 * f32_mat + 2 * bf_act);
    unsigned short* o_bf = (unsigned short*)(p + 3 * f32_mat + 3 * bf_act);
    if (ws_size < bf_act * 5 + bf_w * 4 + f32_mat * 3) return;

    const dim3 blk(256);
    const dim3 gemm_grid(DIM / 128, M / 128);

    // 0) one-shot f32 -> bf16 conversion (bandwidth-bound)
    const size_t n_act = (size_t)M * DIM, n_w = (size_t)DIM * DIM;
    cvt_f32_bf16<<<(unsigned)(n_act / (256 * 8)), blk, 0, stream>>>(x,  x_bf,  n_act);
    cvt_f32_bf16<<<(unsigned)(n_w   / (256 * 8)), blk, 0, stream>>>(Wq, wq_bf, n_w);
    cvt_f32_bf16<<<(unsigned)(n_w   / (256 * 8)), blk, 0, stream>>>(Wk, wk_bf, n_w);
    cvt_f32_bf16<<<(unsigned)(n_w   / (256 * 8)), blk, 0, stream>>>(Wv, wv_bf, n_w);
    cvt_f32_bf16<<<(unsigned)(n_w   / (256 * 8)), blk, 0, stream>>>(Wo, wo_bf, n_w);

    // 1) QKV projections
    gemm_bf16<<<gemm_grid, blk, 0, stream>>>(x_bf, wq_bf, q_f32, M);
    gemm_bf16<<<gemm_grid, blk, 0, stream>>>(x_bf, wk_bf, k_f32, M);
    gemm_bf16<<<gemm_grid, blk, 0, stream>>>(x_bf, wv_bf, v_f32, M);

    // 2) RMSNorm + RoPE (score scale folded into Q)
    const int rows = S * B * H;                       // 65536
    rmsnorm_rope<<<rows / 8, blk, 0, stream>>>(q_f32, gq, rope, q_bf,
                                               0.08838834764831845f, S, B, H);
    rmsnorm_rope<<<rows / 8, blk, 0, stream>>>(k_f32, gk, rope, k_bf, 1.0f, S, B, H);

    // 3) V transpose to (B,H,D,S) bf16
    v_transpose_bf16<<<(unsigned)(n_act / 256), blk, 0, stream>>>(v_f32, vT, S, B, H);

    // 4) flash attention (bf16 output)
    flash_attn<<<dim3(B * H, S / 128), blk, 0, stream>>>(q_bf, k_bf, vT, o_bf, S, B, H);

    // 5) output projection
    gemm_bf16<<<gemm_grid, blk, 0, stream>>>(o_bf, wo_bf, out, M);
}
